// SelfAttHeadLevel3_51969104281925
// MI455X (gfx1250) — matrix operand order repaired and checked by reading, not compiled, not run on hardware
//
#include <hip/hip_runtime.h>
#include <hip/hip_bf16.h>
#include <math.h>

typedef __attribute__((ext_vector_type(16))) __bf16 v16bf;
typedef __attribute__((ext_vector_type(8)))  __bf16 v8bf;
typedef __attribute__((ext_vector_type(8)))  float  v8f;

#define EMB   1024
#define DK    128
#define SEQ   4096
#define BATCH 4

// ---------------------------------------------------------------------------
// Projection GEMM:  out(bf16) = A[M,1024] @ W[1024,NTOT]
// MODE 0: out[M*NTOT + n]                       (QK buffer, NTOT=256)
// MODE 1: out[(b*NTOT + n)*SEQ + s], M=b*S+s    (V stored transposed per batch)
// Block tile 128x64, K-step 32, double-buffered LDS, 8 waves = 4x2 grid of
// 32x32 wave tiles (2x2 WMMA each). Compile-time NTOT -> immediate-offset
// epilogue stores (no per-element branching / 64-bit muls).
// ---------------------------------------------------------------------------
template <int MODE, int NTOT>
__global__ __launch_bounds__(256) void proj_gemm(const float* __restrict__ A,
                                                 const float* __restrict__ W,
                                                 __bf16* __restrict__ out) {
  __shared__ __bf16 As[2][128][40];   // 32 + 8 pad (row stride 80B, 16B aligned)
  __shared__ __bf16 Bs[2][64][40];    // stored transposed: Bs[buf][n][k]

  const int tid  = threadIdx.x;
  const int wave = tid >> 5, lane = tid & 31;
  const int half = lane >> 4, l15 = lane & 15;
  const int wm = wave >> 1, wn = wave & 1;          // 4x2 wave grid
  const int mBase = blockIdx.y * 128;
  const int nBase = blockIdx.x * 64;

  v8f acc[2][2] = {};

  const int arow = tid >> 1;          // 0..127
  const int akc  = (tid & 1) * 16;    // 0 / 16
  const int bn   = tid >> 2;          // 0..63
  const int bks  = (tid & 3) * 8;     // 0,8,16,24

  const float* aSrcBase = A + (size_t)(mBase + arow) * EMB + akc;
  const float* wSrcBase = W + (size_t)bks * NTOT + nBase + bn;

  auto stage = [&](int buf, int k0) {
    // ---- A tile (f32 -> bf16), 16 floats per thread ----
    const float* src = aSrcBase + k0;
    float4 f0 = *(const float4*)(src + 0);
    float4 f1 = *(const float4*)(src + 4);
    float4 f2 = *(const float4*)(src + 8);
    float4 f3 = *(const float4*)(src + 12);
    v8bf p0, p1;
    p0[0]=(__bf16)f0.x; p0[1]=(__bf16)f0.y; p0[2]=(__bf16)f0.z; p0[3]=(__bf16)f0.w;
    p0[4]=(__bf16)f1.x; p0[5]=(__bf16)f1.y; p0[6]=(__bf16)f1.z; p0[7]=(__bf16)f1.w;
    p1[0]=(__bf16)f2.x; p1[1]=(__bf16)f2.y; p1[2]=(__bf16)f2.z; p1[3]=(__bf16)f2.w;
    p1[4]=(__bf16)f3.x; p1[5]=(__bf16)f3.y; p1[6]=(__bf16)f3.z; p1[7]=(__bf16)f3.w;
    *(v8bf*)&As[buf][arow][akc]     = p0;
    *(v8bf*)&As[buf][arow][akc + 8] = p1;
    // ---- B tile transposed: Bs[n][k] = W[(k0+k)*NTOT + nBase+n] ----
    const float* wsrc = wSrcBase + (size_t)k0 * NTOT;
    v8bf pb;
#pragma unroll
    for (int i = 0; i < 8; ++i) pb[i] = (__bf16)wsrc[(size_t)i * NTOT];
    *(v8bf*)&Bs[buf][bn][bks] = pb;
    // prefetch two K-tiles ahead into GL2 (global_prefetch_b8)
    if (k0 + 64 < EMB) {
      __builtin_prefetch(aSrcBase + k0 + 64, 0, 3);
      __builtin_prefetch(wSrcBase + (size_t)(k0 + 64) * NTOT, 0, 3);
    }
  };

  stage(0, 0);
  __syncthreads();

  int buf = 0;
  for (int k0 = 0; k0 < EMB; k0 += 32, buf ^= 1) {
    if (k0 + 32 < EMB) stage(buf ^ 1, k0 + 32);   // overlap with WMMAs below

    v16bf af[2], bf[2];
#pragma unroll
    for (int i = 0; i < 2; ++i) {
      const int row = wm * 32 + i * 16 + l15;
      const v8bf lo = *(const v8bf*)&As[buf][row][half * 8];
      const v8bf hi = *(const v8bf*)&As[buf][row][16 + half * 8];
#pragma unroll
      for (int e = 0; e < 8; ++e) { af[i][e] = lo[e]; af[i][e + 8] = hi[e]; }
    }
#pragma unroll
    for (int j = 0; j < 2; ++j) {
      const int col = wn * 32 + j * 16 + l15;
      const v8bf lo = *(const v8bf*)&Bs[buf][col][half * 8];
      const v8bf hi = *(const v8bf*)&Bs[buf][col][16 + half * 8];
#pragma unroll
      for (int e = 0; e < 8; ++e) { bf[j][e] = lo[e]; bf[j][e + 8] = hi[e]; }
    }
#pragma unroll
    for (int i = 0; i < 2; ++i)
#pragma unroll
      for (int j = 0; j < 2; ++j)
        acc[i][j] = __builtin_amdgcn_wmma_f32_16x16x32_bf16(
            false, af[i], false, bf[j], (short)0, acc[i][j], false, false);
    __syncthreads();
  }

  // ---- epilogue: all offsets compile-time immediates ----
  if (MODE == 0) {
    __bf16* base = out + (size_t)(mBase + wm * 32 + half * 8) * NTOT
                       + nBase + wn * 32 + l15;
#pragma unroll
    for (int i = 0; i < 2; ++i)
#pragma unroll
      for (int j = 0; j < 2; ++j)
#pragma unroll
        for (int r = 0; r < 8; ++r)
          base[(i * 16 + r) * NTOT + j * 16] = (__bf16)acc[i][j][r];
  } else {
    const int bIdx  = mBase >> 12;                       // batch (S=4096)
    const int sBase = (mBase & (SEQ - 1)) + wm * 32 + half * 8;
    __bf16* base = out + ((size_t)bIdx * NTOT + nBase + wn * 32 + l15) * SEQ
                       + sBase;
#pragma unroll
    for (int i = 0; i < 2; ++i)
#pragma unroll
      for (int j = 0; j < 2; ++j)
#pragma unroll
        for (int r = 0; r < 8; ++r)
          base[(size_t)(j * 16) * SEQ + i * 16 + r] = (__bf16)acc[i][j][r];
  }
}

// ---------------------------------------------------------------------------
// Flash attention: one 16-query tile per workgroup (8 waves, 256 threads).
// Key blocks of 256; each wave computes a 32-key score slice cooperatively,
// online softmax in LDS, then each wave owns a 128-wide d_v slice of P@V.
// QK buffer: [b][s][256] bf16 (Q cols 0..127, K cols 128..255).
// Vt buffer: [b][dv][s]  bf16 (transposed -> contiguous B-operand loads).
// ---------------------------------------------------------------------------
__global__ __launch_bounds__(256) void attn_kernel(const __bf16* __restrict__ QK,
                                                   const __bf16* __restrict__ Vt,
                                                   const int* __restrict__ mask,
                                                   float* __restrict__ out) {
  __shared__ float  S_lds[16][264];   // 256 + 8 pad
  __shared__ __bf16 P_lds[16][264];
  __shared__ float  pmax[16][16];
  __shared__ float  psum[16][16];
  __shared__ float  m_lds[16], l_lds[16], f_lds[16];

  const int tid  = threadIdx.x;
  const int wave = tid >> 5, lane = tid & 31;
  const int half = lane >> 4, l15 = lane & 15;
  const int b     = blockIdx.y;
  const int qbase = blockIdx.x * 16;

  const __bf16* QKb = QK + (size_t)b * SEQ * 256;
  const __bf16* Vb  = Vt + (size_t)b * EMB * SEQ;

  // Q fragments for this 16-row tile (4 k-steps of 32 over d_k=128)
  v16bf qf[4];
  {
    const __bf16* qrow = QKb + (size_t)(qbase + l15) * 256;
#pragma unroll
    for (int ks = 0; ks < 4; ++ks) {
      const v8bf lo = *(const v8bf*)(qrow + ks * 32 + half * 8);
      const v8bf hi = *(const v8bf*)(qrow + ks * 32 + 16 + half * 8);
#pragma unroll
      for (int e = 0; e < 8; ++e) { qf[ks][e] = lo[e]; qf[ks][e + 8] = hi[e]; }
    }
  }

  v8f o[8] = {};   // 16 x 128 output slice per wave

  if (tid < 16) { m_lds[tid] = -INFINITY; l_lds[tid] = 0.f; }
  __syncthreads();

  const float scale = 0.08838834764831845f;  // 1/sqrt(128)

  for (int it = 0; it < SEQ / 256; ++it) {
    const int k0 = it * 256;

    // prefetch next key block's K rows (one row per thread) and V chunk
    if (it + 1 < SEQ / 256) {
      __builtin_prefetch(QKb + (size_t)(k0 + 256 + tid) * 256 + DK, 0, 3);
      __builtin_prefetch(Vb + (size_t)(tid * 4) * SEQ + k0 + 256, 0, 3);
    }

    // ---- scores: this wave handles keys [wave*32, wave*32+32) ----
#pragma unroll
    for (int nt = 0; nt < 2; ++nt) {
      v8f s = {};
      const int kc = wave * 32 + nt * 16 + l15;
      const __bf16* krow = QKb + (size_t)(k0 + kc) * 256 + DK;
#pragma unroll
      for (int ks = 0; ks < 4; ++ks) {
        v16bf kf;
        const v8bf lo = *(const v8bf*)(krow + ks * 32 + half * 8);
        const v8bf hi = *(const v8bf*)(krow + ks * 32 + 16 + half * 8);
#pragma unroll
        for (int e = 0; e < 8; ++e) { kf[e] = lo[e]; kf[e + 8] = hi[e]; }
        s = __builtin_amdgcn_wmma_f32_16x16x32_bf16(
            false, qf[ks], false, kf, (short)0, s, false, false);
      }
#pragma unroll
      for (int r = 0; r < 8; ++r) {
        const int row = r + half * 8;
        const int qg  = qbase + row;
        const int kg  = k0 + kc;
        float v = s[r] * scale;
        if (mask[(size_t)qg * SEQ + kg] == 0) v = -INFINITY;
        S_lds[row][kc] = v;
      }
    }
    __syncthreads();

    // ---- online softmax: block max ----
    {
      const int row = tid >> 4, seg = tid & 15;
      float mx = -INFINITY;
#pragma unroll
      for (int c = 0; c < 16; ++c) mx = fmaxf(mx, S_lds[row][seg * 16 + c]);
      pmax[row][seg] = mx;
    }
    __syncthreads();
    if (tid < 16) {
      float bm = -INFINITY;
#pragma unroll
      for (int s2 = 0; s2 < 16; ++s2) bm = fmaxf(bm, pmax[tid][s2]);
      const float mo = m_lds[tid];
      const float nm = fmaxf(mo, bm);
      f_lds[tid] = (mo == -INFINITY) ? 0.f : __expf(mo - nm);
      m_lds[tid] = nm;
    }
    __syncthreads();

    // ---- p = exp(s - m), write bf16 P tile, partial sums ----
    {
      const int row = tid >> 4, seg = tid & 15;
      const float nm = m_lds[row];
      float sum = 0.f;
#pragma unroll
      for (int c = 0; c < 16; ++c) {
        const float p = __expf(S_lds[row][seg * 16 + c] - nm);
        P_lds[row][seg * 16 + c] = (__bf16)p;
        sum += p;
      }
      psum[row][seg] = sum;
    }
    __syncthreads();
    if (tid < 16) {
      float sum = 0.f;
#pragma unroll
      for (int s2 = 0; s2 < 16; ++s2) sum += psum[tid][s2];
      l_lds[tid] = l_lds[tid] * f_lds[tid] + sum;
    }

    // ---- rescale accumulators by per-row factor ----
#pragma unroll
    for (int r = 0; r < 8; ++r) {
      const float fr = f_lds[r + half * 8];
#pragma unroll
      for (int j = 0; j < 8; ++j) o[j][r] *= fr;
    }

    // ---- P @ V : 8 k-steps x 8 n-tiles for this wave's 128-wide dv slice ----
#pragma unroll
    for (int ks = 0; ks < 8; ++ks) {
      v16bf pf;
      {
        const __bf16* prow = &P_lds[l15][ks * 32];
        const v8bf lo = *(const v8bf*)(prow + half * 8);
        const v8bf hi = *(const v8bf*)(prow + 16 + half * 8);
#pragma unroll
        for (int e = 0; e < 8; ++e) { pf[e] = lo[e]; pf[e + 8] = hi[e]; }
      }
#pragma unroll
      for (int j = 0; j < 8; ++j) {
        const __bf16* vrow =
            Vb + (size_t)(wave * 128 + j * 16 + l15) * SEQ + k0 + ks * 32 + half * 8;
        v16bf vf;
        const v8bf lo = *(const v8bf*)(vrow);
        const v8bf hi = *(const v8bf*)(vrow + 16);
#pragma unroll
        for (int e = 0; e < 8; ++e) { vf[e] = lo[e]; vf[e + 8] = hi[e]; }
        o[j] = __builtin_amdgcn_wmma_f32_16x16x32_bf16(
            false, pf, false, vf, (short)0, o[j], false, false);
      }
    }
    __syncthreads();
  }

  // ---- finalize: divide by l, write f32 output ----
#pragma unroll
  for (int r = 0; r < 8; ++r) {
    const int row = r + half * 8;
    const float inv = 1.f / l_lds[row];
    const int qg = qbase + row;
    float* orow = out + ((size_t)b * SEQ + qg) * EMB;
#pragma unroll
    for (int j = 0; j < 8; ++j)
      orow[wave * 128 + j * 16 + l15] = o[j][r] * inv;
  }
}

// ---------------------------------------------------------------------------
extern "C" void kernel_launch(void* const* d_in, const int* in_sizes, int n_in,
                              void* d_out, int out_size, void* d_ws, size_t ws_size,
                              hipStream_t stream) {
  const float* E    = (const float*)d_in[0];
  const int*   mask = (const int*)d_in[1];
  const float* WQK  = (const float*)d_in[2];
  const float* WV   = (const float*)d_in[3];
  float* out = (float*)d_out;

  // workspace: QK bf16 [B][S][256] (8 MB) then Vt bf16 [B][1024][S] (32 MB)
  __bf16* QKb = (__bf16*)d_ws;
  __bf16* Vtb = QKb + (size_t)BATCH * SEQ * 256;

  dim3 blk(256);
  // QK = E @ WQK   -> [B*S, 256] bf16
  proj_gemm<0, 256><<<dim3(256 / 64, (BATCH * SEQ) / 128), blk, 0, stream>>>(
      E, WQK, QKb);
  // V  = E @ WV    -> transposed bf16 [B][1024][S]
  proj_gemm<1, EMB><<<dim3(EMB / 64, (BATCH * SEQ) / 128), blk, 0, stream>>>(
      E, WV, Vtb);
  // flash attention
  attn_kernel<<<dim3(SEQ / 16, BATCH), blk, 0, stream>>>(QKb, Vtb, mask, out);
}
